// PET_body_13486197310271
// MI455X (gfx1250) — matrix operand order repaired and checked by reading, not compiled, not run on hardware
//
#include <hip/hip_runtime.h>
#include <stdint.h>

// ---------------------------------------------------------------------------
// PET transformer body for MI455X (gfx1250, wave32, WMMA).
// All heavy GEMMs use v_wmma_f32_16x16x32_bf16 (fp32 accumulate).
// GEMM: 2x2 register tiling per wave -> 2 b128 loads per WMMA issue.
// ---------------------------------------------------------------------------

#define DEV __device__ __forceinline__

typedef unsigned short u16;
typedef __attribute__((ext_vector_type(16))) __bf16 v16bf;
typedef __attribute__((ext_vector_type(8)))  float  v8f;

union Frag {
  v16bf v;
  uint4 q[2];
  u16   s[16];
};

DEV u16 f2b(float x) {                       // fp32 -> bf16, round-nearest-even
  uint32_t u = __float_as_uint(x);
  uint32_t r = u + 0x7FFFu + ((u >> 16) & 1u);
  return (u16)(r >> 16);
}
DEV float gelu_f(float x) {                  // tanh-approx gelu (jax default)
  float t = tanhf(0.7978845608028654f * (x + 0.044715f * x * x * x));
  return 0.5f * x * (1.0f + t);
}
DEV float silu_f(float x) { return x / (1.0f + __expf(-x)); }

DEV v8f wmma_bf16(const Frag& a, const Frag& b, v8f c) {
  return __builtin_amdgcn_wmma_f32_16x16x32_bf16(false, a.v, false, b.v,
                                                 (short)0, c, false, false);
}

// ---------------------------------------------------------------------------
// Weight prep: W (Kd x Nd fp32, row major) -> WT (Nd x KP bf16, zero-padded K)
// ---------------------------------------------------------------------------
__global__ void k_wt(const float* __restrict__ W, u16* __restrict__ WT,
                     int Kd, int Nd, int KP) {
  int g = blockIdx.x * 256 + threadIdx.x;
  if (g >= Nd * KP) return;
  int n = g / KP, k = g % KP;
  float v = (k < Kd) ? W[(size_t)k * Nd + n] : 0.0f;
  WT[g] = f2b(v);
}

// X (rows x Kd fp32) -> Y (rows x KP bf16, zero padded)
__global__ void k_cvt_pad(const float* __restrict__ X, u16* __restrict__ Y,
                          int rows, int Kd, int KP) {
  int g = blockIdx.x * 256 + threadIdx.x;
  if (g >= rows * KP) return;
  int r = g / KP, k = g % KP;
  Y[g] = f2b(k < Kd ? X[(size_t)r * Kd + k] : 0.0f);
}

// ---------------------------------------------------------------------------
// Generic bf16 WMMA GEMM:  C[M,N] = act(A[M,KP] @ BT[N,KP]^T + bias)
// A row-major bf16 (K padded to mult of 32), BT is B transposed (N x KP).
// 256 threads = 8 waves; each wave owns a 32x32 region = 2x2 WMMA tiles.
// Block tile: 64 rows x 128 cols (waves arranged 2 x 4).
// Fragment layouts follow CDNA5 ISA §7.12.2:
//   A lane: row = lane%16, K = {8h..8h+7, 16+8h..23+8h} per 32-chunk (h=lane/16)
//   B lane: col = lane%16, K = {8h..8h+7, 16+8h..23+8h} (same as A)
//   C lane: col = lane%16, rows = 8h+r in VGPR r
// ---------------------------------------------------------------------------
__global__ __launch_bounds__(256)
void k_gemm(const u16* __restrict__ A, const u16* __restrict__ BT,
            const float* __restrict__ bias, float* __restrict__ Cf,
            u16* __restrict__ Cb, int M, int N, int KP, int act) {
  const int lane = threadIdx.x & 31;
  const int wave = threadIdx.x >> 5;
  const int wr = wave & 1;                   // 2 row groups of 32
  const int wc = wave >> 1;                  // 4 col groups of 32
  const int row0 = blockIdx.y * 64 + wr * 32;
  const int col0 = blockIdx.x * 128 + wc * 32;
  const int m16 = lane & 15, half = lane >> 4;

  v8f acc00 = {0.f,0.f,0.f,0.f,0.f,0.f,0.f,0.f};
  v8f acc01 = acc00, acc10 = acc00, acc11 = acc00;

  const u16* ap0 = A  + (size_t)(row0 + m16) * KP + 8 * half;
  const u16* ap1 = ap0 + (size_t)16 * KP;
  const u16* bp0 = BT + (size_t)(col0 + m16) * KP + 8 * half;
  const u16* bp1 = bp0 + (size_t)16 * KP;

  for (int k0 = 0; k0 < KP; k0 += 32) {
    __builtin_prefetch(ap0 + 256, 0, 1);     // global_prefetch_b8 (L2 warm)
    __builtin_prefetch(bp0 + 256, 0, 1);
    Frag a0, a1, b0, b1;
    a0.q[0] = *(const uint4*)(ap0); a0.q[1] = *(const uint4*)(ap0 + 16);
    a1.q[0] = *(const uint4*)(ap1); a1.q[1] = *(const uint4*)(ap1 + 16);
    b0.q[0] = *(const uint4*)(bp0); b0.q[1] = *(const uint4*)(bp0 + 16);
    b1.q[0] = *(const uint4*)(bp1); b1.q[1] = *(const uint4*)(bp1 + 16);
    ap0 += 32; ap1 += 32; bp0 += 32; bp1 += 32;
    acc00 = wmma_bf16(a0, b0, acc00);
    acc01 = wmma_bf16(a0, b1, acc01);
    acc10 = wmma_bf16(a1, b0, acc10);
    acc11 = wmma_bf16(a1, b1, acc11);
  }

  const int n0 = col0 + m16;
  const int n1 = n0 + 16;
  const float bv0 = bias ? bias[n0] : 0.0f;
  const float bv1 = bias ? bias[n1] : 0.0f;
#pragma unroll
  for (int r = 0; r < 8; ++r) {
    int mA = row0 + 8 * half + r;            // row tile 0
    int mB = mA + 16;                        // row tile 1
    float v00 = acc00[r] + bv0, v01 = acc01[r] + bv1;
    float v10 = acc10[r] + bv0, v11 = acc11[r] + bv1;
    if (act == 1) { v00 = gelu_f(v00); v01 = gelu_f(v01);
                    v10 = gelu_f(v10); v11 = gelu_f(v11); }
    else if (act == 2) { v00 = silu_f(v00); v01 = silu_f(v01);
                         v10 = silu_f(v10); v11 = silu_f(v11); }
    size_t oA = (size_t)mA * N + n0;
    size_t oB = (size_t)mB * N + n0;
    if (Cf) { Cf[oA] = v00; Cf[oA + 16] = v01; Cf[oB] = v10; Cf[oB + 16] = v11; }
    if (Cb) { Cb[oA] = f2b(v00); Cb[oA + 16] = f2b(v01);
              Cb[oB] = f2b(v10); Cb[oB + 16] = f2b(v11); }
  }
}

// ---------------------------------------------------------------------------
// Fused attention: one block per (head, batch). N=128 tokens, HEAD=32.
// Scores + P@V via WMMA; softmax in registers with wave32 shuffles.
// ---------------------------------------------------------------------------
__global__ __launch_bounds__(256)
void k_attn(const u16* __restrict__ qkv, const float* __restrict__ mask,
            u16* __restrict__ attn) {
  const int h = blockIdx.x, bi = blockIdx.y;
  __shared__ u16 q_s[128 * 32];    // 8 KB
  __shared__ u16 k_s[128 * 32];    // 8 KB
  __shared__ u16 v_t[32 * 128];    // 8 KB (V transposed: [dim][token])
  __shared__ u16 p_s[128 * 128];   // 32 KB softmax probs (bf16)
  const int tid = threadIdx.x;

  for (int e = tid; e < 128 * 32; e += 256) {
    int row = e >> 5, c = e & 31;
    size_t base = (size_t)(bi * 128 + row) * 768 + h * 32 + c;
    q_s[e] = qkv[base];
    k_s[e] = qkv[base + 256];
    v_t[c * 128 + row] = qkv[base + 512];
  }
  __syncthreads();

  const int lane = tid & 31, wave = tid >> 5;
  const int m16 = lane & 15, half = lane >> 4;
  const int row0 = wave * 16;                 // 8 waves x 16 rows = 128

  // S = Q @ K^T  (8 column tiles, K=32 -> single WMMA each)
  Frag a, b;
  a.q[0] = *(const uint4*)&q_s[(row0 + m16) * 32 + 8 * half];
  a.q[1] = *(const uint4*)&q_s[(row0 + m16) * 32 + 8 * half + 16];
  v8f sc[8];
#pragma unroll
  for (int c = 0; c < 8; ++c) {
    b.q[0] = *(const uint4*)&k_s[(c * 16 + m16) * 32 + 8 * half];
    b.q[1] = *(const uint4*)&k_s[(c * 16 + m16) * 32 + 8 * half + 16];
    v8f z = {0.f, 0.f, 0.f, 0.f, 0.f, 0.f, 0.f, 0.f};
    sc[c] = wmma_bf16(a, b, z);
  }

  // mask + scale
  float msk[8];
#pragma unroll
  for (int c = 0; c < 8; ++c) msk[c] = mask[bi * 128 + c * 16 + m16];
  const float scl = 0.17677669529663687f;     // 1/sqrt(32)
#pragma unroll
  for (int c = 0; c < 8; ++c)
#pragma unroll
    for (int r = 0; r < 8; ++r)
      sc[c][r] = (msk[c] > 0.0f) ? sc[c][r] * scl : -1.0e9f;

  // softmax per row (row m held by 16 lanes of one half across sc[c][r])
#pragma unroll
  for (int r = 0; r < 8; ++r) {
    float mx = -3.4e38f;
#pragma unroll
    for (int c = 0; c < 8; ++c) mx = fmaxf(mx, sc[c][r]);
    for (int o = 1; o < 16; o <<= 1) mx = fmaxf(mx, __shfl_xor(mx, o, 32));
    float ev[8], sum = 0.0f;
#pragma unroll
    for (int c = 0; c < 8; ++c) { ev[c] = __expf(sc[c][r] - mx); sum += ev[c]; }
    for (int o = 1; o < 16; o <<= 1) sum += __shfl_xor(sum, o, 32);
    float inv = 1.0f / sum;
    int row = row0 + 8 * half + r;
#pragma unroll
    for (int c = 0; c < 8; ++c)
      p_s[row * 128 + c * 16 + m16] = f2b(ev[c] * inv);
  }
  __syncthreads();

  // O = P @ V  (2 col tiles of 16, K=128 -> 4 WMMA steps each)
#pragma unroll
  for (int ct = 0; ct < 2; ++ct) {
    v8f o = {0.f, 0.f, 0.f, 0.f, 0.f, 0.f, 0.f, 0.f};
    for (int k0 = 0; k0 < 128; k0 += 32) {
      a.q[0] = *(const uint4*)&p_s[(row0 + m16) * 128 + k0 + 8 * half];
      a.q[1] = *(const uint4*)&p_s[(row0 + m16) * 128 + k0 + 8 * half + 16];
      b.q[0] = *(const uint4*)&v_t[(ct * 16 + m16) * 128 + k0 + 8 * half];
      b.q[1] = *(const uint4*)&v_t[(ct * 16 + m16) * 128 + k0 + 8 * half + 16];
      o = wmma_bf16(a, b, o);
    }
    int n = ct * 16 + m16;
#pragma unroll
    for (int r = 0; r < 8; ++r) {
      int row = row0 + 8 * half + r;
      attn[(size_t)(bi * 128 + row) * 256 + h * 32 + n] = f2b(o[r]);
    }
  }
}

// ---------------------------------------------------------------------------
// LayerNorm epilogue: x = LN(res + hin*[gamma]*[mask]) * g + b  (D = 256)
// Writes fp32 x (in place allowed) and bf16 copy.
// ---------------------------------------------------------------------------
__global__ __launch_bounds__(256)
void k_ln(const float* __restrict__ res, const float* __restrict__ hin,
          const float* __restrict__ gamma, const float* __restrict__ mask,
          const float* __restrict__ g, const float* __restrict__ bta,
          float* __restrict__ xout, u16* __restrict__ xbf) {
  int t = blockIdx.x, d = threadIdx.x;
  size_t o = (size_t)t * 256 + d;
  float hv = hin[o];
  if (gamma) hv *= gamma[d];
  if (mask)  hv *= mask[t];
  float v = res[o] + hv;
  __shared__ float s1[256], s2[256];
  s1[d] = v; s2[d] = v * v;
  __syncthreads();
  for (int s = 128; s > 0; s >>= 1) {
    if (d < s) { s1[d] += s1[d + s]; s2[d] += s2[d + s]; }
    __syncthreads();
  }
  float mean = s1[0] * (1.0f / 256.0f);
  float var  = s2[0] * (1.0f / 256.0f) - mean * mean;
  float out = (v - mean) * rsqrtf(var + 1e-5f) * g[d] + bta[d];
  xout[o] = out;
  if (xbf) xbf[o] = f2b(out);
}

// ---------------------------------------------------------------------------
// Fourier time embedding -> bf16 (B=128 rows x 256)
// ---------------------------------------------------------------------------
__global__ void k_fourier(const float* __restrict__ tin, u16* __restrict__ e) {
  int b = blockIdx.x, i = threadIdx.x;       // 128 threads
  float t = tin[b];
  float emb = 9.210340371976184f / 127.0f;   // log(1e4)/(half-1)
  float freq = __expf(-emb * (float)i);
  float ang = t * freq * 1000.0f;
  e[b * 256 + i]       = f2b(sinf(ang));
  e[b * 256 + 128 + i] = f2b(cosf(ang));
}

// enc' = enc*(1+mask*scale) + mask*shift + lf ; -> x, skip, x_bf
__global__ void k_combine(const float* __restrict__ enc, const float* __restrict__ ts,
                          const float* __restrict__ mask, const float* __restrict__ lf,
                          float* __restrict__ x, float* __restrict__ skip,
                          u16* __restrict__ xbf) {
  int g = blockIdx.x * 256 + threadIdx.x;
  int t = g >> 8, d = g & 255;
  int b = t >> 7;
  float m = mask[t];
  float v = enc[g] * (1.0f + m * ts[b * 512 + d]) + m * ts[b * 512 + 256 + d] + lf[g];
  x[g] = v; skip[g] = v; xbf[g] = f2b(v);
}

__global__ void k_final(const float* __restrict__ x, const float* __restrict__ skip,
                        float* __restrict__ out) {
  int g = blockIdx.x * 256 + threadIdx.x;
  out[g] = x[g] + skip[g];
}

// ---------------------------------------------------------------------------
// KNN support
// ---------------------------------------------------------------------------
__global__ void k_shift(const float* __restrict__ in, const float* __restrict__ mask,
                        float* __restrict__ out, int Pd, int total) {
  int g = blockIdx.x * 256 + threadIdx.x;
  if (g >= total) return;
  out[g] = in[g] + 999.0f * (1.0f - mask[g / Pd]);
}

__global__ void k_rownorm(const float* __restrict__ pts, float* __restrict__ nrm, int Pd) {
  int t = blockIdx.x * 256 + threadIdx.x;   // T total threads (exact)
  const float* p = pts + (size_t)t * Pd;
  float s = 0.0f;
  for (int i = 0; i < Pd; ++i) s += p[i] * p[i];
  nrm[t] = s;
}

// block per (n, b): 128 threads compute distances, thread 0 selects 11 nearest
__global__ void k_knn(const float* __restrict__ pts, const float* __restrict__ nrm,
                      int Pd, int* __restrict__ idx) {
  int b = blockIdx.y, n = blockIdx.x, m = threadIdx.x;
  __shared__ float dist[128];
  __shared__ int taken[128];
  const float* pn = pts + (size_t)(b * 128 + n) * Pd;
  const float* pm = pts + (size_t)(b * 128 + m) * Pd;
  float dot = 0.0f;
  for (int p = 0; p < Pd; ++p) dot += pn[p] * pm[p];
  dist[m] = nrm[b * 128 + n] + nrm[b * 128 + m] - 2.0f * dot;
  taken[m] = 0;
  __syncthreads();
  if (m == 0) {
    for (int j = 0; j < 11; ++j) {          // K+1 picks, skip the first (self)
      float bd = 3.4e38f; int bix = 0;
      for (int q = 0; q < 128; ++q)
        if (!taken[q] && dist[q] < bd) { bd = dist[q]; bix = q; }
      taken[bix] = 1;
      if (j > 0) idx[(size_t)(b * 128 + n) * 10 + (j - 1)] = bix;
    }
  }
}

// rows = 8 batches * 128 tokens * 10 neighbors; pair row = [nbr-ctr | ctr | 0pad]
__global__ void k_gather(const float* __restrict__ X, const int* __restrict__ idx,
                         u16* __restrict__ pair, int b0, int Fd, int KP) {
  int row = blockIdx.x;
  int bl = row / 1280, rem = row % 1280;
  int n = rem / 10, j = rem % 10;
  int t  = (b0 + bl) * 128 + n;
  int tn = (b0 + bl) * 128 + idx[(size_t)t * 10 + j];
  const float* xc = X + (size_t)t  * Fd;
  const float* xn = X + (size_t)tn * Fd;
  u16* pr = pair + (size_t)row * KP;
  for (int c = threadIdx.x; c < KP; c += blockDim.x) {
    float v;
    if (c < Fd)          v = xn[c] - xc[c];
    else if (c < 2 * Fd) v = xc[c - Fd];
    else                 v = 0.0f;
    pr[c] = f2b(v);
  }
}

__global__ void k_meank(const float* __restrict__ outl, float* __restrict__ lf, int b0) {
  int g = blockIdx.x * 256 + threadIdx.x;   // 8*128*256 exact
  int tl = g >> 8, d = g & 255;
  float s = 0.0f;
  for (int j = 0; j < 10; ++j) s += outl[(size_t)(tl * 10 + j) * 256 + d];
  lf[(size_t)(b0 * 128 + tl) * 256 + d] = s * 0.1f;
}

// ---------------------------------------------------------------------------
// Host orchestration
// ---------------------------------------------------------------------------
extern "C" void kernel_launch(void* const* d_in, const int* in_sizes, int n_in,
                              void* d_out, int out_size, void* d_ws, size_t ws_size,
                              hipStream_t stream) {
  (void)in_sizes; (void)n_in; (void)out_size; (void)ws_size;
  const int T = 16384;  // B*N

  const float* input_features = (const float*)d_in[0];
  const float* input_points   = (const float*)d_in[1];
  const float* input_mask     = (const float*)d_in[2];
  const float* input_time     = (const float*)d_in[3];
  const float* enc_w1  = (const float*)d_in[4];
  const float* enc_b1  = (const float*)d_in[5];
  const float* enc_w2  = (const float*)d_in[6];
  const float* enc_b2  = (const float*)d_in[7];
  const float* four_w1 = (const float*)d_in[8];
  const float* four_w2 = (const float*)d_in[9];
  const float* time_w  = (const float*)d_in[10];
  const float* loc1_w1 = (const float*)d_in[11];
  const float* loc1_b1 = (const float*)d_in[12];
  const float* loc1_w2 = (const float*)d_in[13];
  const float* loc1_b2 = (const float*)d_in[14];
  const float* loc2_w1 = (const float*)d_in[15];
  const float* loc2_b1 = (const float*)d_in[16];
  const float* loc2_w2 = (const float*)d_in[17];
  const float* loc2_b2 = (const float*)d_in[18];
  const float* qkv_w   = (const float*)d_in[19];
  const float* qkv_b   = (const float*)d_in[20];
  const float* out_w   = (const float*)d_in[21];
  const float* out_b   = (const float*)d_in[22];
  const float* ln1_g   = (const float*)d_in[23];
  const float* ln1_b   = (const float*)d_in[24];
  const float* ff1_w   = (const float*)d_in[25];
  const float* ff1_b   = (const float*)d_in[26];
  const float* ff2_w   = (const float*)d_in[27];
  const float* ff2_b   = (const float*)d_in[28];
  const float* gammaW  = (const float*)d_in[29];
  const float* ln2_g   = (const float*)d_in[30];
  const float* ln2_b   = (const float*)d_in[31];

  char* ws = (char*)d_ws;
  size_t off = 0;
  auto alloc = [&](size_t bytes) -> char* {
    char* p = ws + off;
    off += (bytes + 255) & ~(size_t)255;
    return p;
  };
  // bf16 transposed weights
  u16* encW1T  = (u16*)alloc(512 * 32 * 2);
  u16* encW2T  = (u16*)alloc(256 * 512 * 2);
  u16* fourW1T = (u16*)alloc(512 * 256 * 2);
  u16* fourW2T = (u16*)alloc(256 * 512 * 2);
  u16* timeWT  = (u16*)alloc(512 * 256 * 2);
  u16* loc1W1T = (u16*)alloc(512 * 32 * 2);
  u16* loc1W2T = (u16*)alloc(256 * 512 * 2);
  u16* loc2W1T = (u16*)alloc(512 * 512 * 2);
  u16* loc2W2T = (u16*)alloc(256 * 512 * 2);
  u16* qkvWT   = (u16*)alloc((size_t)8 * 768 * 256 * 2);
  u16* outWT   = (u16*)alloc((size_t)8 * 256 * 256 * 2);
  u16* ff1WT   = (u16*)alloc((size_t)8 * 512 * 256 * 2);
  u16* ff2WT   = (u16*)alloc((size_t)8 * 256 * 512 * 2);
  // activations
  u16*   featP = (u16*)alloc((size_t)T * 32 * 2);
  u16*   e1    = (u16*)alloc((size_t)T * 512 * 2);
  float* encB  = (float*)alloc((size_t)T * 256 * 4);
  u16*   eFour = (u16*)alloc((size_t)128 * 256 * 2);
  u16*   h4    = (u16*)alloc((size_t)128 * 512 * 2);
  u16*   tEmb  = (u16*)alloc((size_t)128 * 256 * 2);
  float* ts    = (float*)alloc((size_t)128 * 512 * 4);
  float* pts1  = (float*)alloc((size_t)T * 2 * 4);
  float* nrmB  = (float*)alloc((size_t)T * 4);
  int*   idx   = (int*)alloc((size_t)T * 10 * 4);
  float* lf1   = (float*)alloc((size_t)T * 256 * 4);
  float* lf2   = (float*)alloc((size_t)T * 256 * 4);
  u16*   pair  = (u16*)alloc((size_t)10240 * 512 * 2);
  u16*   hidl  = (u16*)alloc((size_t)10240 * 512 * 2);
  float* outl  = (float*)alloc((size_t)10240 * 256 * 4);
  float* x     = (float*)alloc((size_t)T * 256 * 4);
  float* skip  = (float*)alloc((size_t)T * 256 * 4);
  u16*   xbf   = (u16*)alloc((size_t)T * 256 * 2);
  u16*   qkvA  = (u16*)alloc((size_t)T * 768 * 2);
  u16*   attnB = (u16*)alloc((size_t)T * 256 * 2);
  u16*   hid   = (u16*)alloc((size_t)T * 512 * 2);
  float* tmp   = (float*)alloc((size_t)T * 256 * 4);
  float* pts2  = tmp;   // alias: only used before the transformer loop

  auto wprep = [&](const float* W, u16* WT, int Kd, int Nd, int KP) {
    int tot = Nd * KP;
    k_wt<<<(tot + 255) / 256, 256, 0, stream>>>(W, WT, Kd, Nd, KP);
  };
  auto gemm = [&](const u16* A, const u16* BT, const float* bias, float* Cf,
                  u16* Cb, int M, int N, int KP, int act) {
    dim3 g(N / 128, M / 64);                 // block tile 64 x 128
    k_gemm<<<g, 256, 0, stream>>>(A, BT, bias, Cf, Cb, M, N, KP, act);
  };

  // ---- weight prep ----
  wprep(enc_w1,  encW1T,  13,  512, 32);
  wprep(enc_w2,  encW2T,  512, 256, 512);
  wprep(four_w1, fourW1T, 256, 512, 256);
  wprep(four_w2, fourW2T, 512, 256, 512);
  wprep(time_w,  timeWT,  256, 512, 256);
  wprep(loc1_w1, loc1W1T, 26,  512, 32);
  wprep(loc1_w2, loc1W2T, 512, 256, 512);
  wprep(loc2_w1, loc2W1T, 512, 512, 512);
  wprep(loc2_w2, loc2W2T, 512, 256, 512);
  for (int i = 0; i < 8; ++i) {
    wprep(qkv_w + (size_t)i * 256 * 768, qkvWT + (size_t)i * 768 * 256, 256, 768, 256);
    wprep(out_w + (size_t)i * 256 * 256, outWT + (size_t)i * 256 * 256, 256, 256, 256);
    wprep(ff1_w + (size_t)i * 256 * 512, ff1WT + (size_t)i * 512 * 256, 256, 512, 256);
    wprep(ff2_w + (size_t)i * 512 * 256, ff2WT + (size_t)i * 256 * 512, 512, 256, 512);
  }

  // ---- encoder MLP ----
  k_cvt_pad<<<(T * 32 + 255) / 256, 256, 0, stream>>>(input_features, featP, T, 13, 32);
  gemm(featP, encW1T, enc_b1, nullptr, e1, T, 512, 32, 1);
  gemm(e1, encW2T, enc_b2, encB, nullptr, T, 256, 512, 1);

  // ---- Fourier time conditioning ----
  k_fourier<<<128, 128, 0, stream>>>(input_time, eFour);
  gemm(eFour, fourW1T, nullptr, nullptr, h4, 128, 512, 256, 2);
  gemm(h4, fourW2T, nullptr, nullptr, tEmb, 128, 256, 512, 2);
  gemm(tEmb, timeWT, nullptr, ts, nullptr, 128, 512, 256, 0);

  // ---- KNN local features, pass 1 (2-D points, 13-D feats) ----
  k_shift<<<(T * 2 + 255) / 256, 256, 0, stream>>>(input_points, input_mask, pts1, 2, T * 2);
  k_rownorm<<<T / 256, 256, 0, stream>>>(pts1, nrmB, 2);
  k_knn<<<dim3(128, 128), 128, 0, stream>>>(pts1, nrmB, 2, idx);
  for (int c = 0; c < 16; ++c) {
    int b0 = c * 8;
    k_gather<<<10240, 64, 0, stream>>>(input_features, idx, pair, b0, 13, 32);
    gemm(pair, loc1W1T, loc1_b1, nullptr, hidl, 10240, 512, 32, 1);
    gemm(hidl, loc1W2T, loc1_b2, outl, nullptr, 10240, 256, 512, 1);
    k_meank<<<1024, 256, 0, stream>>>(outl, lf1, b0);
  }

  // ---- KNN local features, pass 2 (256-D points == lf1) ----
  k_shift<<<(T * 256 + 255) / 256, 256, 0, stream>>>(lf1, input_mask, pts2, 256, T * 256);
  k_rownorm<<<T / 256, 256, 0, stream>>>(pts2, nrmB, 256);
  k_knn<<<dim3(128, 128), 128, 0, stream>>>(pts2, nrmB, 256, idx);
  for (int c = 0; c < 16; ++c) {
    int b0 = c * 8;
    k_gather<<<10240, 256, 0, stream>>>(lf1, idx, pair, b0, 256, 512);
    gemm(pair, loc2W1T, loc2_b1, nullptr, hidl, 10240, 512, 512, 1);
    gemm(hidl, loc2W2T, loc2_b2, outl, nullptr, 10240, 256, 512, 1);
    k_meank<<<1024, 256, 0, stream>>>(outl, lf2, b0);
  }

  // ---- combine: time-conditioned enc + local features -> x, skip ----
  k_combine<<<T, 256, 0, stream>>>(encB, ts, input_mask, lf2, x, skip, xbf);

  // ---- transformer layers ----
  for (int i = 0; i < 8; ++i) {
    gemm(xbf, qkvWT + (size_t)i * 768 * 256, qkv_b + (size_t)i * 768,
         nullptr, qkvA, T, 768, 256, 0);
    k_attn<<<dim3(8, 128), 256, 0, stream>>>(qkvA, input_mask, attnB);
    gemm(attnB, outWT + (size_t)i * 256 * 256, out_b + (size_t)i * 256,
         tmp, nullptr, T, 256, 256, 0);
    k_ln<<<T, 256, 0, stream>>>(x, tmp, nullptr, nullptr,
                                ln1_g + (size_t)i * 256, ln1_b + (size_t)i * 256, x, xbf);
    gemm(xbf, ff1WT + (size_t)i * 512 * 256, ff1_b + (size_t)i * 512,
         nullptr, hid, T, 512, 256, 1);
    gemm(hid, ff2WT + (size_t)i * 256 * 512, ff2_b + (size_t)i * 256,
         tmp, nullptr, T, 256, 512, 0);
    k_ln<<<T, 256, 0, stream>>>(x, tmp, gammaW + (size_t)i * 256, input_mask,
                                ln2_g + (size_t)i * 256, ln2_b + (size_t)i * 256, x, xbf);
  }

  // ---- output: x + skip ----
  k_final<<<T, 256, 0, stream>>>(x, skip, (float*)d_out);
}
